// CrossAttentionBlock_57183194579609
// MI455X (gfx1250) — matrix operand order repaired and checked by reading, not compiled, not run on hardware
//
#include <hip/hip_runtime.h>
#include <hip/hip_bf16.h>
#include <math.h>

typedef __attribute__((ext_vector_type(16))) _Float16 v16h;
typedef __attribute__((ext_vector_type(8)))  _Float16 v8h;
typedef __attribute__((ext_vector_type(4)))  _Float16 v4h;
typedef __attribute__((ext_vector_type(8)))  float    v8f;
typedef __attribute__((ext_vector_type(4)))  float    v4f;

enum { Bb = 4, Nn = 2048, Mm = 1024, Cc = 1024, Hh = 16, HDd = 64, HIDd = 4096 };
#define SCALE 0.125f   /* 1/sqrt(64) */

#define BM 128
#define BN 128
#define BK 32

__device__ __forceinline__ v16h cat16(v8h lo, v8h hi) {
    return __builtin_shufflevector(lo, hi, 0,1,2,3,4,5,6,7,8,9,10,11,12,13,14,15);
}
// 16 contiguous halves (32B)
__device__ __forceinline__ v16h ld_frag_contig(const _Float16* p) {
    return cat16(*(const v8h*)p, *(const v8h*)(p + 8));
}
// A-layout fragment: halves {kb..kb+7, kb+16..kb+23}
__device__ __forceinline__ v16h ld_frag_split(const _Float16* p) {
    return cat16(*(const v8h*)p, *(const v8h*)(p + 16));
}

// ---------------------------------------------------------------------------
// Tiled WMMA GEMM, double-buffered LDS: out[M,N] = A[M,K] @ W[K,N] (+ epilogue)
// MODE 0: out = acc (+bias if non-null)
// MODE 1: out = ls[n]*(acc + bias[n]) + res[m,n]
// MODE 2: out = ls[n]*((acc + bias[n]) * tanh(gate[n])) + res[m,n]
// ---------------------------------------------------------------------------
template <int MODE>
__global__ __launch_bounds__(256)
void gemm_wmma(const float* __restrict__ A, const float* __restrict__ W,
               const float* __restrict__ bias, const float* __restrict__ ls,
               const float* __restrict__ res, const float* __restrict__ gate,
               float* __restrict__ out, int Mrows, int Kdim, int Ncols)
{
    alignas(16) __shared__ _Float16 Alds[2][BM * BK];   // [row][k]
    alignas(16) __shared__ _Float16 Blds[2][BN * BK];   // transposed: [n][k]

    const int tid   = threadIdx.x;
    const int lane  = tid & 31;
    const int wid   = tid >> 5;
    const int waveM = wid & 3;    // 4 waves along M (32 rows each)
    const int waveN = wid >> 2;   // 2 waves along N (64 cols each)

    const int bm = blockIdx.y * BM;
    const int bn = blockIdx.x * BN;

    v8f acc[2][4];
#pragma unroll
    for (int mi = 0; mi < 2; ++mi)
#pragma unroll
        for (int ni = 0; ni < 4; ++ni)
            acc[mi][ni] = (v8f)(0.0f);

    // stage one K-tile (f32 -> f16) into LDS buffer `buf`
    auto stage = [&](int buf, int kk) {
        _Float16* As = Alds[buf];
        _Float16* Bs = Blds[buf];
#pragma unroll
        for (int it = 0; it < 4; ++it) {
            int idx = it * 256 + tid;          // 0..1023, 4 elems each
            int row = idx >> 3;                // 0..127
            int col = (idx & 7) << 2;          // 0..28
            v4f a4 = *(const v4f*)(A + (size_t)(bm + row) * Kdim + kk + col);
            v4h h4; h4[0] = (_Float16)a4[0]; h4[1] = (_Float16)a4[1];
            h4[2] = (_Float16)a4[2]; h4[3] = (_Float16)a4[3];
            *(v4h*)(As + row * BK + col) = h4;
        }
#pragma unroll
        for (int it = 0; it < 4; ++it) {
            int idx  = it * 256 + tid;
            int krow = idx >> 5;               // 0..31
            int ncol = (idx & 31) << 2;        // 0..124
            v4f b4 = *(const v4f*)(W + (size_t)(kk + krow) * Ncols + bn + ncol);
#pragma unroll
            for (int j = 0; j < 4; ++j)
                Bs[(ncol + j) * BK + krow] = (_Float16)b4[j];
        }
    };

    stage(0, 0);
    int cur = 0;

    for (int kk = 0; kk < Kdim; kk += BK) {
        __syncthreads();                        // staging of `cur` complete
        if (kk + BK < Kdim) {
            // light 2-ahead prefetch into cache (global_prefetch_b8)
            if (kk + 2 * BK < Kdim) {
                __builtin_prefetch(A + (size_t)(bm + (tid >> 1)) * Kdim + kk + 2 * BK + (tid & 1) * 16, 0, 3);
                if (tid < 32)
                    __builtin_prefetch(W + (size_t)(kk + 2 * BK + tid) * Ncols + bn, 0, 3);
            }
            stage(cur ^ 1, kk + BK);            // overlap with compute below
        }

        const _Float16* As = Alds[cur];
        const _Float16* Bs = Blds[cur];
        v16h af[2], bf[4];
        {
            const int r0 = waveM * 32 + (lane & 15);
            const int kb = (lane >> 4) * 8;
#pragma unroll
            for (int mi = 0; mi < 2; ++mi)
                af[mi] = ld_frag_split(As + (r0 + mi * 16) * BK + kb);
        }
        {
            const int n0 = waveN * 64 + (lane & 15);
            const int kb = (lane >> 4) * 8;
#pragma unroll
            for (int ni = 0; ni < 4; ++ni)
                bf[ni] = ld_frag_split(Bs + (n0 + ni * 16) * BK + kb);
        }
#pragma unroll
        for (int mi = 0; mi < 2; ++mi)
#pragma unroll
            for (int ni = 0; ni < 4; ++ni)
                acc[mi][ni] = __builtin_amdgcn_wmma_f32_16x16x32_f16(
                    false, af[mi], false, bf[ni], (short)0, acc[mi][ni], false, false);
        cur ^= 1;
    }

    // ---- epilogue: C/D layout: lane<16 -> col=lane,row=j ; lane>=16 -> row=8+j
    const int rbase = bm + waveM * 32 + ((lane >> 4) ? 8 : 0);
    const int cbase = bn + waveN * 64 + (lane & 15);
#pragma unroll
    for (int mi = 0; mi < 2; ++mi) {
#pragma unroll
        for (int ni = 0; ni < 4; ++ni) {
            const int n = cbase + ni * 16;
            float bv = 0.f, lsn = 0.f, gn = 0.f;
            if (MODE == 0) { if (bias) bv = bias[n]; }
            else           { bv = bias[n]; lsn = ls[n]; }
            if (MODE == 2) gn = tanhf(gate[n]);
#pragma unroll
            for (int j = 0; j < 8; ++j) {
                const int m = rbase + mi * 16 + j;
                float v = acc[mi][ni][j];
                float o;
                if (MODE == 0)      o = v + bv;
                else if (MODE == 1) o = lsn * (v + bv) + res[(size_t)m * Ncols + n];
                else                o = lsn * ((v + bv) * gn) + res[(size_t)m * Ncols + n];
                out[(size_t)m * Ncols + n] = o;
            }
        }
    }
}

// ---------------------------------------------------------------------------
// Per-head LayerNorm + RoPE prep, f32 [rows, rowStride] head-slice ->
// f16 [B, H, Nrows, HD].  One wave per (row, head); HD = 64 = 2 per lane.
// flags: bit0 = LayerNorm, bit1 = RoPE
// ---------------------------------------------------------------------------
__global__ __launch_bounds__(256)
void head_prep(const float* __restrict__ in, int rowStride,
               _Float16* __restrict__ out,
               const float* __restrict__ g, const float* __restrict__ bta,
               const float* __restrict__ rope, int Nrows, int flags)
{
    const int unit = blockIdx.x * 8 + (threadIdx.x >> 5);  // (b*Nrows + n)*H + h
    const int lane = threadIdx.x & 31;
    const int h    = unit & (Hh - 1);
    const int row  = unit >> 4;          // b*Nrows + n
    const int n    = row % Nrows;
    const int b    = row / Nrows;

    const float* p = in + (size_t)row * rowStride + h * HDd;
    float x0 = p[lane], x1 = p[lane + 32];

    if (flags & 1) {
        float s = x0 + x1;
#pragma unroll
        for (int off = 16; off; off >>= 1) s += __shfl_xor(s, off, 32);
        const float mean = s * (1.0f / 64.0f);
        const float d0 = x0 - mean, d1 = x1 - mean;
        float vs = d0 * d0 + d1 * d1;
#pragma unroll
        for (int off = 16; off; off >>= 1) vs += __shfl_xor(vs, off, 32);
        const float r = rsqrtf(vs * (1.0f / 64.0f) + 1e-6f);
        x0 = d0 * r * g[lane]      + bta[lane];
        x1 = d1 * r * g[lane + 32] + bta[lane + 32];
    }
    if (flags & 2) {
        const float* rp = rope + (size_t)n * (2 * HDd);
        const float s0 = rp[lane],      s1 = rp[lane + 32];
        const float c0 = rp[64 + lane], c1 = rp[64 + lane + 32];
        // rot[2i] = -x[2i+1]; rot[2i+1] = x[2i]  -> partner lives in lane^1
        const float p0 = __shfl_xor(x0, 1, 32);
        const float p1 = __shfl_xor(x1, 1, 32);
        const float r0 = (lane & 1) ? p0 : -p0;
        const float r1 = (lane & 1) ? p1 : -p1;
        x0 = x0 * c0 + r0 * s0;
        x1 = x1 * c1 + r1 * s1;
    }
    _Float16* q = out + ((size_t)(b * Hh + h) * Nrows + n) * HDd;
    q[lane]      = (_Float16)x0;
    q[lane + 32] = (_Float16)x1;
}

// ---------------------------------------------------------------------------
// WMMA flash attention, double-buffered K/V tiles.
// Q/K/V: f16 [B,H,Nx,HD]; out: f32 [B,Nq,H,HD].
// Block = 8 waves = 128 queries of one (b,h). Wave owns 16 query rows.
// Key blocks of 64: S = Q@K^T (8 WMMA), online softmax in C-layout,
// P transposed via per-wave LDS, O += P@V (8 WMMA, f32 accum).
// ---------------------------------------------------------------------------
#define SKB 64
#define LDK 72   // padded LDS row stride (halves): 144B, 16B aligned, conflict-free

__global__ __launch_bounds__(256)
void attn_wmma(const _Float16* __restrict__ Q, const _Float16* __restrict__ K,
               const _Float16* __restrict__ V, float* __restrict__ out,
               int Nq, int Nk)
{
    alignas(16) __shared__ _Float16 Klds[2][SKB * LDK];      // [key][d]
    alignas(16) __shared__ _Float16 Vtlds[2][HDd * LDK];     // [d][key]
    alignas(16) __shared__ _Float16 Plds[8 * 16 * LDK];      // per-wave [qrow][key]

    const int tid  = threadIdx.x;
    const int lane = tid & 31;
    const int wid  = tid >> 5;
    const int bh   = blockIdx.y;
    const int b    = bh >> 4;
    const int h    = bh & 15;
    const int q0   = blockIdx.x * 128 + wid * 16;

    const _Float16* Qb = Q + (size_t)bh * Nq * HDd;
    const _Float16* Kb = K + (size_t)bh * Nk * HDd;
    const _Float16* Vb = V + (size_t)bh * Nk * HDd;

    // persistent Q fragments: kf selects d = kf*32 .. kf*32+31
    v16h qf[2];
    {
        const _Float16* qrow = Qb + (size_t)(q0 + (lane & 15)) * HDd;
        const int kb = (lane >> 4) * 8;
#pragma unroll
        for (int kf = 0; kf < 2; ++kf)
            qf[kf] = ld_frag_split(qrow + kf * 32 + kb);
    }

    float mrow[8], lrow[8];
    v8f oacc[4];
#pragma unroll
    for (int j = 0; j < 8; ++j) { mrow[j] = -1e30f; lrow[j] = 0.0f; }
#pragma unroll
    for (int dn = 0; dn < 4; ++dn) oacc[dn] = (v8f)(0.0f);

    _Float16* Pw = Plds + wid * 16 * LDK;

    // stage K [key][d] and V^T [d][key] (64 keys x 64 dims) into buffer `buf`
    auto stage = [&](int buf, int kb0) {
        _Float16* Ks = Klds[buf];
        _Float16* Vs = Vtlds[buf];
#pragma unroll
        for (int c = 0; c < 2; ++c) {
            const int vecidx = tid * 2 + c;      // 0..511
            const int key = vecidx >> 3;         // 0..63
            const int d0  = (vecidx & 7) * 8;    // 0..56
            v8h kv = *(const v8h*)(Kb + (size_t)(kb0 + key) * HDd + d0);
            *(v8h*)(Ks + key * LDK + d0) = kv;
            v8h vv = *(const v8h*)(Vb + (size_t)(kb0 + key) * HDd + d0);
#pragma unroll
            for (int i = 0; i < 8; ++i)
                Vs[(d0 + i) * LDK + key] = vv[i];
        }
    };

    stage(0, 0);
    int cur = 0;

    for (int kb0 = 0; kb0 < Nk; kb0 += SKB) {
        __syncthreads();                         // staging of `cur` complete
        if (kb0 + SKB < Nk)
            stage(cur ^ 1, kb0 + SKB);           // overlap with compute below

        const _Float16* Ks = Klds[cur];
        const _Float16* Vs = Vtlds[cur];

        // ---- S = scale * Q @ K^T -------------------------------------------
        v8f sacc[4];
#pragma unroll
        for (int nt = 0; nt < 4; ++nt) sacc[nt] = (v8f)(0.0f);
#pragma unroll
        for (int kf = 0; kf < 2; ++kf) {
#pragma unroll
            for (int nt = 0; nt < 4; ++nt) {
                v16h bf = ld_frag_split(
                    Ks + (nt * 16 + (lane & 15)) * LDK + (lane >> 4) * 8 + kf * 32);
                sacc[nt] = __builtin_amdgcn_wmma_f32_16x16x32_f16(
                    false, qf[kf], false, bf, (short)0, sacc[nt], false, false);
            }
        }
#pragma unroll
        for (int nt = 0; nt < 4; ++nt)
#pragma unroll
            for (int j = 0; j < 8; ++j) sacc[nt][j] *= SCALE;

        // ---- online softmax (rows j live across the 16-lane half) ----------
        float rmax[8];
#pragma unroll
        for (int j = 0; j < 8; ++j)
            rmax[j] = fmaxf(fmaxf(sacc[0][j], sacc[1][j]), fmaxf(sacc[2][j], sacc[3][j]));
#pragma unroll
        for (int off = 1; off < 16; off <<= 1)
#pragma unroll
            for (int j = 0; j < 8; ++j) rmax[j] = fmaxf(rmax[j], __shfl_xor(rmax[j], off, 32));

        float sc[8];
#pragma unroll
        for (int j = 0; j < 8; ++j) {
            const float nm = fmaxf(mrow[j], rmax[j]);
            sc[j] = __expf(mrow[j] - nm);
            mrow[j] = nm;
        }
        float rsum[8];
#pragma unroll
        for (int j = 0; j < 8; ++j) rsum[j] = 0.0f;
#pragma unroll
        for (int nt = 0; nt < 4; ++nt)
#pragma unroll
            for (int j = 0; j < 8; ++j) {
                const float pv = __expf(sacc[nt][j] - mrow[j]);
                sacc[nt][j] = pv;
                rsum[j] += pv;
            }
#pragma unroll
        for (int off = 1; off < 16; off <<= 1)
#pragma unroll
            for (int j = 0; j < 8; ++j) rsum[j] += __shfl_xor(rsum[j], off, 32);
#pragma unroll
        for (int j = 0; j < 8; ++j) lrow[j] = lrow[j] * sc[j] + rsum[j];
#pragma unroll
        for (int dn = 0; dn < 4; ++dn)
#pragma unroll
            for (int j = 0; j < 8; ++j) oacc[dn][j] *= sc[j];

        // ---- transpose P (C-layout -> A-layout) via per-wave LDS -----------
        {
            const int prow = (lane >> 4) ? 8 : 0;
#pragma unroll
            for (int nt = 0; nt < 4; ++nt)
#pragma unroll
                for (int j = 0; j < 8; ++j)
                    Pw[(prow + j) * LDK + nt * 16 + (lane & 15)] = (_Float16)sacc[nt][j];
        }
        // same-wave DS ordering: compiler inserts s_wait_dscnt, no barrier needed

        // ---- O += P @ V -----------------------------------------------------
#pragma unroll
        for (int kf = 0; kf < 2; ++kf) {
            v16h pf = ld_frag_split(Pw + (lane & 15) * LDK + (lane >> 4) * 8 + kf * 32);
#pragma unroll
            for (int dn = 0; dn < 4; ++dn) {
                v16h bf = ld_frag_split(
                    Vs + (dn * 16 + (lane & 15)) * LDK + (lane >> 4) * 8 + kf * 32);
                oacc[dn] = __builtin_amdgcn_wmma_f32_16x16x32_f16(
                    false, pf, false, bf, (short)0, oacc[dn], false, false);
            }
        }
        cur ^= 1;
    }

    // ---- normalize + store: out[b, q, h, d] --------------------------------
    const int qrbase = q0 + ((lane >> 4) ? 8 : 0);
    const int dcol   = lane & 15;
#pragma unroll
    for (int j = 0; j < 8; ++j) {
        const float inv = 1.0f / lrow[j];
        float* orow = out + ((size_t)(b * Nq + qrbase + j) * Hh + h) * HDd;
#pragma unroll
        for (int dn = 0; dn < 4; ++dn)
            orow[dn * 16 + dcol] = oacc[dn][j] * inv;
    }
}

// ---------------------------------------------------------------------------
// h = silu(g) * t   (written back over g)
// ---------------------------------------------------------------------------
__global__ __launch_bounds__(256)
void silu_mul(const float* __restrict__ g, const float* __restrict__ t,
              float* __restrict__ h, size_t n)
{
    const size_t i = (size_t)blockIdx.x * blockDim.x + threadIdx.x;
    if (i < n) {
        const float gv = g[i];
        h[i] = (gv / (1.0f + __expf(-gv))) * t[i];
    }
}

// ---------------------------------------------------------------------------
extern "C" void kernel_launch(void* const* d_in, const int* in_sizes, int n_in,
                              void* d_out, int out_size, void* d_ws, size_t ws_size,
                              hipStream_t stream)
{
    const float* x       = (const float*)d_in[0];
    const float* ctx     = (const float*)d_in[1];
    const float* rope    = (const float*)d_in[2];
    const float* wqkv    = (const float*)d_in[3];
    const float* sa_qg   = (const float*)d_in[4];
    const float* sa_qb   = (const float*)d_in[5];
    const float* sa_kg   = (const float*)d_in[6];
    const float* sa_kb   = (const float*)d_in[7];
    const float* sa_wo   = (const float*)d_in[8];
    const float* sa_bo   = (const float*)d_in[9];
    const float* ca_wq   = (const float*)d_in[10];
    const float* ca_wk   = (const float*)d_in[11];
    const float* ca_wv   = (const float*)d_in[12];
    const float* ca_qg   = (const float*)d_in[13];
    const float* ca_qb   = (const float*)d_in[14];
    const float* ca_kg   = (const float*)d_in[15];
    const float* ca_kb   = (const float*)d_in[16];
    const float* ca_wo   = (const float*)d_in[17];
    const float* ca_bo   = (const float*)d_in[18];
    const float* ca_gate = (const float*)d_in[19];
    const float* w1g     = (const float*)d_in[20];
    const float* b1g     = (const float*)d_in[21];
    const float* w1x     = (const float*)d_in[22];
    const float* b1x     = (const float*)d_in[23];
    const float* w2      = (const float*)d_in[24];
    const float* b2      = (const float*)d_in[25];
    const float* ls0     = (const float*)d_in[26];
    const float* ls1     = (const float*)d_in[27];
    const float* ls2     = (const float*)d_in[28];
    float* outp = (float*)d_out;

    char* w = (char*)d_ws;
    const size_t MB = 1ull << 20;
    // region 0 (0..128MB): qkv_raw(96MB) -> ca raw(64MB) -> hg(128MB)
    float*    qkv_raw = (float*)(w);
    float*    caq_raw = (float*)(w);
    float*    cak_raw = (float*)(w + 32 * MB);
    float*    cav_raw = (float*)(w + 48 * MB);
    float*    hg      = (float*)(w);
    float*    hx      = (float*)(w + 128 * MB);
    _Float16* f16q    = (_Float16*)(w + 256 * MB);
    _Float16* f16k    = (_Float16*)(w + 272 * MB);
    _Float16* f16v    = (_Float16*)(w + 288 * MB);
    float*    attn_o  = (float*)(w + 304 * MB);
    float*    x1      = (float*)(w + 336 * MB);   // total ws use: 368 MB

    const int rowsX = Bb * Nn;   // 8192
    const int rowsC = Bb * Mm;   // 4096
    dim3 blk(256);

    // 1. qkv = x @ wqkv
    gemm_wmma<0><<<dim3(3 * Cc / BN, rowsX / BM), blk, 0, stream>>>(
        x, wqkv, nullptr, nullptr, nullptr, nullptr, qkv_raw, rowsX, Cc, 3 * Cc);

    // 2. SA head prep
    const int unitsN = Bb * Nn * Hh;   // 131072
    const int unitsM = Bb * Mm * Hh;   // 65536
    head_prep<<<unitsN / 8, blk, 0, stream>>>(qkv_raw,          3 * Cc, f16q, sa_qg, sa_qb, rope, Nn, 3);
    head_prep<<<unitsN / 8, blk, 0, stream>>>(qkv_raw + Cc,     3 * Cc, f16k, sa_kg, sa_kb, rope, Nn, 3);
    head_prep<<<unitsN / 8, blk, 0, stream>>>(qkv_raw + 2 * Cc, 3 * Cc, f16v, nullptr, nullptr, nullptr, Nn, 0);

    // 3. self-attention (WMMA flash)
    attn_wmma<<<dim3(Nn / 128, Bb * Hh), blk, 0, stream>>>(f16q, f16k, f16v, attn_o, Nn, Nn);

    // 4. x1 = ls0*(attn @ sa_wo + sa_bo) + x
    gemm_wmma<1><<<dim3(Cc / BN, rowsX / BM), blk, 0, stream>>>(
        attn_o, sa_wo, sa_bo, ls0, x, nullptr, x1, rowsX, Cc, Cc);

    // 5-7. cross-attention projections
    gemm_wmma<0><<<dim3(Cc / BN, rowsX / BM), blk, 0, stream>>>(
        x1, ca_wq, nullptr, nullptr, nullptr, nullptr, caq_raw, rowsX, Cc, Cc);
    gemm_wmma<0><<<dim3(Cc / BN, rowsC / BM), blk, 0, stream>>>(
        ctx, ca_wk, nullptr, nullptr, nullptr, nullptr, cak_raw, rowsC, Cc, Cc);
    gemm_wmma<0><<<dim3(Cc / BN, rowsC / BM), blk, 0, stream>>>(
        ctx, ca_wv, nullptr, nullptr, nullptr, nullptr, cav_raw, rowsC, Cc, Cc);

    // 8. CA head prep: q LN+RoPE; k LN only; v plain
    head_prep<<<unitsN / 8, blk, 0, stream>>>(caq_raw, Cc, f16q, ca_qg, ca_qb, rope, Nn, 3);
    head_prep<<<unitsM / 8, blk, 0, stream>>>(cak_raw, Cc, f16k, ca_kg, ca_kb, nullptr, Mm, 1);
    head_prep<<<unitsM / 8, blk, 0, stream>>>(cav_raw, Cc, f16v, nullptr, nullptr, nullptr, Mm, 0);

    // 9. cross-attention (WMMA flash)
    attn_wmma<<<dim3(Nn / 128, Bb * Hh), blk, 0, stream>>>(f16q, f16k, f16v, attn_o, Nn, Mm);

    // 10. x2 (in d_out) = ls1*((attn @ ca_wo + ca_bo) * tanh(gate)) + x1
    gemm_wmma<2><<<dim3(Cc / BN, rowsX / BM), blk, 0, stream>>>(
        attn_o, ca_wo, ca_bo, ls1, x1, ca_gate, outp, rowsX, Cc, Cc);

    // 11-12. MLP up-projections
    gemm_wmma<0><<<dim3(HIDd / BN, rowsX / BM), blk, 0, stream>>>(
        outp, w1g, b1g, nullptr, nullptr, nullptr, hg, rowsX, Cc, HIDd);
    gemm_wmma<0><<<dim3(HIDd / BN, rowsX / BM), blk, 0, stream>>>(
        outp, w1x, b1x, nullptr, nullptr, nullptr, hx, rowsX, Cc, HIDd);

    // 13. h = silu(hg) * hx
    const size_t nh = (size_t)rowsX * HIDd;
    silu_mul<<<(unsigned)((nh + 255) / 256), blk, 0, stream>>>(hg, hx, hg, nh);

    // 14. out = ls2*(h @ w2 + b2) + x2
    gemm_wmma<1><<<dim3(Cc / BN, rowsX / BM), blk, 0, stream>>>(
        hg, w2, b2, ls2, outp, nullptr, outp, rowsX, HIDd, Cc);
}
